// Conv2Fromer_30030411334030
// MI455X (gfx1250) — matrix operand order repaired and checked by reading, not compiled, not run on hardware
//
#include <hip/hip_runtime.h>
#include <math.h>

typedef __attribute__((ext_vector_type(16))) _Float16 v16h;
typedef __attribute__((ext_vector_type(8)))  float    v8f;

#define BB 16
#define CC 64
#define HDIM 64
#define WDIM 96
#define MM 6
#define DD 192
#define NHEAD 8
#define CH 64
#define INNER 512
#define LCHUNK 768
#define XBATCH (CC*HDIM*WDIM)   /* 393216 */
#define DUMP8 (768 * 8)         /* dump slot (8 floats) for pad half-fragment */

// ---------------- Kernel 1: pooled x -> conv -> z update -------------------
// pool(conv(x)) == conv(pool(x)); pooled x is [64 ch][6 pos] per batch.
__global__ __launch_bounds__(384) void k_pool_zadd(const float* __restrict__ x,
                                                   const float* __restrict__ z,
                                                   const float* __restrict__ cw,
                                                   const float* __restrict__ cb,
                                                   float* __restrict__ znew) {
  __shared__ float pool[CC * MM];
  const int b = blockIdx.x, t = threadIdx.x;
  if (t < CC * MM) {
    const int c = t / MM, m = t % MM, p0 = m / 3, p1 = m % 3;
    const float* px = x + b * XBATCH + c * (HDIM * WDIM) + p0 * 32 * WDIM + p1 * 32;
    float s = 0.f;
    for (int r = 0; r < 32; ++r) {
      const float* pr = px + r * WDIM;
      for (int q = 0; q < 32; ++q) s += pr[q];
    }
    pool[c * MM + m] = s * (1.0f / 1024.0f);
  }
  __syncthreads();
  for (int j = t; j < MM * DD; j += 384) {
    const int m = j / DD, d = j % DD;
    float acc = cb[d];
    for (int c = 0; c < CC; ++c) acc += cw[d * CC + c] * pool[c * MM + m];
    znew[(b * MM + m) * DD + d] = z[(b * MM + m) * DD + d] + acc;
  }
}

// ---------------- Kernel 2: q projection with baked-in weird .view ---------
// q[b,hd,mm,ch] = q_lin[b, base>>3, (base&7)*64+ch], base = hd*6+mm
__global__ __launch_bounds__(256) void k_qproj(const float* __restrict__ znew,
                                               const float* __restrict__ wq,
                                               const float* __restrict__ bq,
                                               float* __restrict__ qbuf) {
  const int idx = blockIdx.x * 256 + threadIdx.x;   // 49152 total
  const int b = idx / 3072, r = idx % 3072;
  const int base = r >> 6, ch = r & 63;
  const int m = base >> 3;
  const int col = ((base & 7) << 6) + ch;
  float acc = bq[col];
  const float* zr = znew + (b * MM + m) * DD;
  for (int k = 0; k < DD; ++k) acc += zr[k] * wq[k * INNER + col];
  qbuf[idx] = acc;
}

// ---------------- Kernel 3: attention per (b, head) via WMMA ---------------
// K=V = contiguous 768x64 chunk of x (flat view); repetition-by-8 cancels in
// softmax, so this computes the reference exactly (modulo f16 rounding).
__global__ __launch_bounds__(128) void k_attn(const float* __restrict__ x,
                                              const float* __restrict__ qbuf,
                                              float* __restrict__ oattn) {
  __shared__ _Float16 qa[16 * 64];          // Q (pre-scaled) padded to 16 rows
  __shared__ float    dots_t[768 * 8 + 8];  // logits, [col][row] + dump slot
  __shared__ _Float16 pa[16 * 768];         // softmax probs, rows 6..15 zero
  const int tid = threadIdx.x;
  const int wv = tid >> 5, lane = tid & 31;
  const int b = blockIdx.x >> 3, hd = blockIdx.x & 7;
  const float* xb = x + b * XBATCH + hd * LCHUNK * CH;   // [768][64] row-major

  // stage Q rows 0..5 scaled by 1/sqrt(64), zero-pad rows 6..15,
  // zero only pad rows of pa (rows 0..5 are fully overwritten by softmax)
  for (int i = tid; i < 6 * 64; i += 128)
    qa[i] = (_Float16)(qbuf[(b * 48 + hd * 6) * 64 + i] * 0.125f);
  for (int i = 6 * 64 + tid; i < 16 * 64; i += 128) qa[i] = (_Float16)0.f;
  for (int i = 6 * 768 + tid; i < 16 * 768; i += 128) pa[i] = (_Float16)0.f;
  __syncthreads();

  // A fragments of Q (contiguous K order used here: K in [16h,16h+16), matching B)
  const int am = lane & 15, ah = lane >> 4;
  v16h a0, a1;
  {
    const _Float16* p0 = qa + am * 64 + 16 * ah;
    const _Float16* p1 = qa + am * 64 + 32 + 16 * ah;
    for (int i = 0; i < 8; ++i) {
      a0[i] = p0[i]; a0[8 + i] = p0[8 + i];
      a1[i] = p1[i]; a1[8 + i] = p1[8 + i];
    }
  }

  // dots = (Q*scale) @ K^T ; 48 N-tiles of 16 keys split across 4 waves.
  // B operand (32x16): lane holds column n = lane&15, K sequential from 16h,
  // and B[k][n] = K_row(n)[k] -> contiguous loads from the key row.
  // C store: lane's 8 VGPRs are rows 8*bh+0..7 of column (t*16+bn); the
  // bh=0 half holds all valid rows (0..5) and lands as 8 contiguous floats
  // at dots_t[col*8]; the bh=1 half (pad rows) goes to one dump slot --
  // a single address select per tile, stores merge into 2x b128.
  const int bn = lane & 15, bh = lane >> 4;
  for (int t = wv; t < 48; t += 4) {
    const float* kr = xb + (t * 16 + bn) * CH;
    v16h b0, b1;
    for (int i = 0; i < 16; ++i) {
      b0[i] = (_Float16)kr[16 * bh + i];
      b1[i] = (_Float16)kr[32 + 16 * bh + i];
    }
    v8f cacc = {};
    cacc = __builtin_amdgcn_wmma_f32_16x16x32_f16(false, a0, false, b0,
                                                  (short)0, cacc, false, false);
    cacc = __builtin_amdgcn_wmma_f32_16x16x32_f16(false, a1, false, b1,
                                                  (short)0, cacc, false, false);
    float* dst = &dots_t[(bh == 0) ? ((t * 16 + bn) * 8) : DUMP8];
    for (int i = 0; i < 8; ++i) dst[i] = cacc[i];
  }
  __syncthreads();

  // softmax over 768 per valid row (wave-parallel rows, shfl reductions);
  // single exp pass: stash exp(d-max) in pa, then scale by 1/sum.
  for (int row = wv; row < 6; row += 4) {
    const float* dr = dots_t + row;           // element i at dr[i*8]
    _Float16* pr = pa + row * 768;
    float mx = -3.4e38f;
    for (int i = lane; i < 768; i += 32) mx = fmaxf(mx, dr[i * 8]);
    for (int off = 16; off > 0; off >>= 1) mx = fmaxf(mx, __shfl_xor(mx, off, 32));
    float sum = 0.f;
    for (int i = lane; i < 768; i += 32) {
      const float e = __expf(dr[i * 8] - mx);
      sum += e;
      pr[i] = (_Float16)e;
    }
    for (int off = 16; off > 0; off >>= 1) sum += __shfl_xor(sum, off, 32);
    const float inv = 1.0f / sum;
    for (int i = lane; i < 768; i += 32)
      pr[i] = (_Float16)((float)pr[i] * inv);
  }
  __syncthreads();

  // out = attn @ V : M=16, N-tile = wave id (4x16 = 64 cols), K = 768 (24 steps)
  v8f oc = {};
  for (int ks = 0; ks < 24; ++ks) {
    v16h af, bf;
    const _Float16* ap = pa + am * 768 + ks * 32 + 16 * ah;
    for (int i = 0; i < 8; ++i) { af[i] = ap[i]; af[8 + i] = ap[8 + i]; }
    // B[k][n] = V[k][n]: column n of x chunk, stride 64 floats (coalesced
    // across lanes n=0..15 at each k)
    const float* vp = xb + (ks * 32 + 16 * bh) * CH + wv * 16 + bn;
    for (int i = 0; i < 16; ++i) bf[i] = (_Float16)vp[i * CH];
    oc = __builtin_amdgcn_wmma_f32_16x16x32_f16(false, af, false, bf,
                                                (short)0, oc, false, false);
  }
  for (int i = 0; i < 8; ++i) {
    const int row = i + 8 * bh;
    if (row < 6)
      oattn[(b * MM + row) * INNER + hd * CH + wv * 16 + bn] = oc[i];
  }
}

// ---------------- Kernel 4: output projection + residual -------------------
__global__ __launch_bounds__(256) void k_oproj(const float* __restrict__ znew,
                                               const float* __restrict__ oattn,
                                               const float* __restrict__ wo,
                                               const float* __restrict__ bo,
                                               float* __restrict__ out) {
  const int idx = blockIdx.x * 256 + threadIdx.x;   // 18432 total
  const int d = idx % DD;
  const int bm = idx / DD;
  float acc = znew[idx] + bo[d];
  const float* orow = oattn + bm * INNER;
  for (int k = 0; k < INNER; ++k) acc += orow[k] * wo[k * DD + d];
  out[idx] = acc;
}

extern "C" void kernel_launch(void* const* d_in, const int* in_sizes, int n_in,
                              void* d_out, int out_size, void* d_ws, size_t ws_size,
                              hipStream_t stream) {
  const float* x  = (const float*)d_in[0];
  const float* z  = (const float*)d_in[1];
  const float* cw = (const float*)d_in[2];
  const float* cb = (const float*)d_in[3];
  const float* wq = (const float*)d_in[4];
  const float* bq = (const float*)d_in[5];
  const float* wo = (const float*)d_in[6];
  const float* bo = (const float*)d_in[7];
  float* out  = (float*)d_out;

  float* znew  = (float*)d_ws;          // 16*6*192   = 18432 floats
  float* qbuf  = znew + 18432;          // 16*8*6*64  = 49152 floats
  float* oattn = qbuf + 49152;          // 16*6*512   = 49152 floats

  k_pool_zadd<<<BB, 384, 0, stream>>>(x, z, cw, cb, znew);
  k_qproj<<<192, 256, 0, stream>>>(znew, wq, bq, qbuf);
  k_attn<<<BB * NHEAD, 128, 0, stream>>>(x, qbuf, oattn);
  k_oproj<<<72, 256, 0, stream>>>(znew, oattn, wo, bo, out);
}
